// uLSIF_49581102465423
// MI455X (gfx1250) — matrix operand order repaired and checked by reading, not compiled, not run on hardware
//
#include <hip/hip_runtime.h>
#include <hip/hip_bf16.h>

#define NN 8192
#define DD 128
#define TEMP 128.0f
// LDS B-tile row: 256 data bf16 (128 hi | 128 lo) + 8 pad bf16 (16B) => 528B row
// 528B/4 = 132 banks -> 4-bank rotation per row across the 64 LDS banks.
#define LDS_ROW 264

typedef __attribute__((ext_vector_type(16))) __bf16 v16bf;
typedef __attribute__((ext_vector_type(8)))  float  v8f;
typedef __attribute__((ext_vector_type(4)))  unsigned int v4u;
typedef __attribute__((ext_vector_type(8)))  int v8i;
typedef __attribute__((ext_vector_type(4)))  int v4i;

union Frag {
    v16bf v;
    uint4 q[2];
};

// A fragment (16x32 bf16 A layout): lane L: M = L%16, elem e -> K = k0 + 8*(L/16) + (e<8 ? e : e+8)
__device__ inline v16bf load_fragA(const __bf16* __restrict__ base, int row, int k0, int lane) {
    const __bf16* p = base + (size_t)row * DD + k0 + ((lane >> 4) << 3);
    Frag f;
    f.q[0] = *(const uint4*)(p);        // e = 0..7
    f.q[1] = *(const uint4*)(p + 16);   // e = 8..15 (K offset +16)
    return f.v;
}

// B fragment from LDS-staged tile. Tile row n (= column c0+n) holds 256 bf16:
// [0..127] = hi(K), [128..255] = lo(K), row stride LDS_ROW.
// B layout: lane L: N = L%16, elem e -> K = k0 + 8*(L/16) + (e<8 ? e : e+8)
__device__ inline v16bf load_fragB_lds(const __bf16* buf, int lane, int part, int k0) {
    const __bf16* p = buf + (lane & 15) * LDS_ROW + part * 128 + k0 + ((lane >> 4) << 3);
    Frag f;
    f.q[0] = *(const uint4*)(p);        // e = 0..7
    f.q[1] = *(const uint4*)(p + 16);   // e = 8..15 (K offset +16)
    return f.v;
}

// Tensor Data Mover: load one 16-row x 512B tile of apack into LDS with
// 16B padding every 512B (row stride 528B in LDS). D# per CDNA5 ISA §8.
// 6-arg builtin form (clang-23 / therock headers): g0, g1, g2, g3, extra, cpol.
__device__ inline void tdm_load_tile(unsigned lds_off, unsigned long long ga) {
    // group0: count=1 | lds_addr | global_addr[56:0] | type=2
    v4u g0 = { 1u,
               lds_off,
               (unsigned)ga,
               ((unsigned)((ga >> 32) & 0x01FFFFFFull)) | 0x80000000u };
    // group1: data_size=2B, pad_enable, pad_interval=6 (512B), pad_amount=3 (16B),
    // tensor_dim0=256, tensor_dim1=8192, tile_dim0=256, tile_dim1=16, stride0=256
    v8i g1 = { (int)((1u << 16) | (1u << 20) | (6u << 22) | (3u << 25)),
               (int)(256u << 16),      // [63:48] tensor_dim0[15:0]
               (int)(8192u << 16),     // [95:80] tensor_dim1[15:0]
               (int)(256u << 16),      // [127:112] tile_dim0
               16,                     // [143:128] tile_dim1
               256,                    // [191:160] tensor_dim0_stride[31:0]
               0, 0 };
    v4i z4 = { 0, 0, 0, 0 };
    v8i z8 = { 0, 0, 0, 0, 0, 0, 0, 0 };
    __builtin_amdgcn_tensor_load_to_lds(g0, g1, z4, z4, z8, 0);
}

// Split fp32 -> bf16 hi + residual lo. x goes to split arrays (A side),
// a goes to a packed [row][128 hi | 128 lo] layout for single-descriptor TDM.
__global__ __launch_bounds__(256) void ulsif_cvt_split(
    const float* __restrict__ x, const float* __restrict__ a,
    __bf16* __restrict__ xhi, __bf16* __restrict__ xlo,
    __bf16* __restrict__ apack, int total) {
    int i = blockIdx.x * blockDim.x + threadIdx.x;
    if (i < total) {
        float f = x[i];
        __bf16 h = (__bf16)f;
        xhi[i] = h;
        xlo[i] = (__bf16)(f - (float)h);
        f = a[i];
        h = (__bf16)f;
        int j = i >> 7, k = i & 127;
        apack[j * 256 + k]       = h;
        apack[j * 256 + 128 + k] = (__bf16)(f - (float)h);
    }
}

__global__ __launch_bounds__(256) void ulsif_gemm(
    const __bf16* __restrict__ xhi, const __bf16* __restrict__ xlo,
    const __bf16* __restrict__ apack, float* __restrict__ partials) {
    __shared__ __align__(16) __bf16 bufs[2][16 * LDS_ROW];
    __shared__ float smem[8];

    const int lane = threadIdx.x & 31;
    const int wave = threadIdx.x >> 5;
    const int rb   = blockIdx.x;          // row block: 128 rows
    const int cs   = blockIdx.y;          // column slice: 64 tiles
    const int jtBase = cs * 64;
    const int r0   = rb * 128 + wave * 16;

    // Persistent A fragments: this wave's 16 rows, K = 0..127, hi and lo parts.
    v16bf Ah[4], Al[4];
    const int arow = r0 + (lane & 15);
#pragma unroll
    for (int kb = 0; kb < 4; ++kb) {
        Ah[kb] = load_fragA(xhi, arow, kb * 32, lane);
        Al[kb] = load_fragA(xlo, arow, kb * 32, lane);
    }

    const float invT    = 1.0f / TEMP;
    const float twoInvT = 2.0f / TEMP;
    const float invNm1  = 1.0f / (float)(NN - 1);

    // C layout: vgpr k, lane L -> M = k + 8*(L/16), N = L%16.
    const int  diagK   = (lane & 15) - ((lane >> 4) << 3);
    const bool hasDiag = (diagK >= 0) && (diagK < 8);
    const int  jtDiag  = r0 >> 4;         // this wave's diagonal column tile

    const unsigned long long gbase = (unsigned long long)(uintptr_t)apack;
    const unsigned lds0 = (unsigned)(uintptr_t)(&bufs[0][0]);
    const unsigned lds1 = (unsigned)(uintptr_t)(&bufs[1][0]);

    // Prologue: stage tile 0 via TDM (wave 0 drives the DMA engine).
    if (wave == 0) {
        tdm_load_tile(lds0, gbase + (unsigned long long)jtBase * 8192ull);
    }

    float accS = 0.0f;   // sum of exp(2z/T) over visited elements
    float accD = 0.0f;   // diagonal extras: -2*exp(z/T) - exp(2z/T)/(N-1)

    for (int i = 0; i < 64; ++i) {
        const int jt = jtBase + i;
        if (wave == 0) {
            if (i + 1 < 64) {
                // Prefetch next tile into the other buffer, then wait for the
                // current tile only (in-order TENSORcnt: <=1 means tile i done).
                tdm_load_tile((i & 1) ? lds0 : lds1,
                              gbase + (unsigned long long)(jt + 1) * 8192ull);
                __builtin_amdgcn_s_wait_tensorcnt(1);
            } else {
                __builtin_amdgcn_s_wait_tensorcnt(0);
            }
        }
        __syncthreads();                  // tile i visible to all waves
        asm volatile("" ::: "memory");    // LDS was written by the TDM, not us

        const __bf16* buf = (i & 1) ? bufs[1] : bufs[0];

        // Hoist all 8 B fragments first: one clause of 16 ds_load_b128 and a
        // single dscnt wait, then 12 back-to-back accumulator-chained WMMAs.
        v16bf Bh[4], Bl[4];
#pragma unroll
        for (int kb = 0; kb < 4; ++kb) {
            Bh[kb] = load_fragB_lds(buf, lane, 0, kb * 32);
            Bl[kb] = load_fragB_lds(buf, lane, 1, kb * 32);
        }

        v8f c = {0.f, 0.f, 0.f, 0.f, 0.f, 0.f, 0.f, 0.f};
#pragma unroll
        for (int kb = 0; kb < 4; ++kb) {
            // z = (hi+lo)x . (hi+lo)a ~= hi.hi + lo.hi + hi.lo (fp32 accumulate)
            c = __builtin_amdgcn_wmma_f32_16x16x32_bf16(false, Ah[kb], false, Bh[kb],
                                                        (short)0, c, false, false);
            c = __builtin_amdgcn_wmma_f32_16x16x32_bf16(false, Al[kb], false, Bh[kb],
                                                        (short)0, c, false, false);
            c = __builtin_amdgcn_wmma_f32_16x16x32_bf16(false, Ah[kb], false, Bl[kb],
                                                        (short)0, c, false, false);
        }
        const bool isDiagTile = (jt == jtDiag);
#pragma unroll
        for (int k = 0; k < 8; ++k) {
            float z  = c[k];
            float e2 = __expf(z * twoInvT);   // sim^2 = exp(2z/T)
            accS += e2;
            if (isDiagTile && hasDiag && (k == diagK)) {
                accD += -2.0f * __expf(z * invT) - e2 * invNm1;
            }
        }
        __syncthreads();                  // all reads done before buf is re-filled
    }

    // Deterministic wave32 shuffle reduction, then cross-wave via LDS.
    float v = accS * invNm1 + accD;
#pragma unroll
    for (int m = 16; m >= 1; m >>= 1)
        v += __shfl_xor(v, m, 32);
    if (lane == 0) smem[wave] = v;
    __syncthreads();
    if (threadIdx.x == 0) {
        float s = 0.0f;
#pragma unroll
        for (int w = 0; w < 8; ++w) s += smem[w];
        partials[cs * 64 + rb] = s;
    }
}

__global__ __launch_bounds__(512) void ulsif_reduce(
    const float* __restrict__ partials, float* __restrict__ out, int n) {
    __shared__ float s[512];
    int t = threadIdx.x;
    s[t] = (t < n) ? partials[t] : 0.0f;
    __syncthreads();
#pragma unroll
    for (int step = 256; step > 0; step >>= 1) {
        if (t < step) s[t] += s[t + step];
        __syncthreads();
    }
    if (t == 0) out[0] = s[0] / (float)NN;
}

extern "C" void kernel_launch(void* const* d_in, const int* in_sizes, int n_in,
                              void* d_out, int out_size, void* d_ws, size_t ws_size,
                              hipStream_t stream) {
    const float* x = (const float*)d_in[0];
    const float* a = (const float*)d_in[1];

    char* ws = (char*)d_ws;
    const size_t half = (size_t)NN * DD * sizeof(__bf16);  // 2 MB
    __bf16* xhi   = (__bf16*)(ws);
    __bf16* xlo   = (__bf16*)(ws + half);
    __bf16* apack = (__bf16*)(ws + 2 * half);              // 4 MB packed hi|lo
    float*  partials = (float*)(ws + 4 * half);            // 512 floats

    const int total = NN * DD;
    ulsif_cvt_split<<<(total + 255) / 256, 256, 0, stream>>>(x, a, xhi, xlo, apack, total);
    dim3 grid(NN / 128, 8);   // 64 row blocks x 8 column slices
    ulsif_gemm<<<grid, 256, 0, stream>>>(xhi, xlo, apack, partials);
    ulsif_reduce<<<1, 512, 0, stream>>>(partials, (float*)d_out, NN / 16);
}
